// MultiPeakAwareAttention_80032420593855
// MI455X (gfx1250) — matrix operand order repaired and checked by reading, not compiled, not run on hardware
//
#include <hip/hip_runtime.h>

typedef __attribute__((ext_vector_type(16))) _Float16 v16h;
typedef __attribute__((ext_vector_type(8)))  _Float16 v8h;
typedef __attribute__((ext_vector_type(4)))  _Float16 v4h;
typedef __attribute__((ext_vector_type(8)))  float    v8f;
typedef __attribute__((ext_vector_type(4)))  float    v4f;

#define BB   4
#define NN   4096
#define CC   128
#define HH   64
#define WW   64
#define KTOP 4
#define INV_T 20.0f   // 1 / 0.05

// ---------------------------------------------------------------------------
// K1: L2-normalize rows of (B*N, 128) fp32 and convert to f16.
// One wave (32 lanes) per row, 4 elements per lane, wave32 shfl reduction.
// ---------------------------------------------------------------------------
__global__ __launch_bounds__(256) void norm_cvt_kernel(
    const float* __restrict__ src, _Float16* __restrict__ dst, int rows) {
  int gwave = (blockIdx.x * blockDim.x + threadIdx.x) >> 5;
  int lane  = threadIdx.x & 31;
  if (gwave >= rows) return;
  const float* r = src + (size_t)gwave * CC;
  v4f x = *(const v4f*)(r + lane * 4);
  float ss = x.x * x.x + x.y * x.y + x.z * x.z + x.w * x.w;
#pragma unroll
  for (int off = 16; off > 0; off >>= 1) ss += __shfl_xor(ss, off, 32);
  float sc = 1.0f / fmaxf(sqrtf(ss), 1e-12f);
  v4h o;
  o.x = (_Float16)(x.x * sc); o.y = (_Float16)(x.y * sc);
  o.z = (_Float16)(x.z * sc); o.w = (_Float16)(x.w * sc);
  *(v4h*)(dst + (size_t)gwave * CC + lane * 4) = o;
}

// ---------------------------------------------------------------------------
// K2: sim[b] = Anorm[b] (4096x128) * Bnorm[b]^T (128x4096), fp32 out.
// Block = 4 waves covering a 64(M) x 64(N) tile.  Each wave owns a 16x64
// strip: 4 f32 accumulators, A fragment loaded once per K-step and reused
// across 4 v_wmma_f32_16x16x32_f16 (16 WMMAs / wave).  Fragment k order is
// contiguous k = 16*hi + s in A and B (b128 loads, no transposes needed).
// ---------------------------------------------------------------------------
__global__ __launch_bounds__(128) void sim_gemm_kernel(
    const _Float16* __restrict__ Ah, const _Float16* __restrict__ Bh,
    float* __restrict__ sim) {
  const int lane = threadIdx.x & 31;
  const int wv   = threadIdx.x >> 5;
  const int b    = blockIdx.z;
  const int row0 = (blockIdx.y * 4 + wv) * 16;  // 16-row strip per wave
  const int col0 = blockIdx.x * 64;             // 64-col tile per block
  const int hi   = lane >> 4;                   // 0: lanes 0-15, 1: 16-31
  const int lm   = lane & 15;

  const _Float16* arow = Ah + ((size_t)(b * NN + row0 + lm)) * CC;
  const _Float16* brow = Bh + ((size_t)(b * NN + col0 + lm)) * CC;

  v8f acc0 = {}, acc1 = {}, acc2 = {}, acc3 = {};
#pragma unroll
  for (int k0 = 0; k0 < CC; k0 += 32) {
    // A 16x32: lane m holds K = k0+0..15 (lo half-wave) or k0+16..31 (hi)
    v8h a_lo = *(const v8h*)(arow + k0 + hi * 16);
    v8h a_hi = *(const v8h*)(arow + k0 + 8 + hi * 16);
    v16h a;
#pragma unroll
    for (int i = 0; i < 8; ++i) { a[i] = a_lo[i]; a[i + 8] = a_hi[i]; }
    // B 32x16: lane n holds column (col0 + 16*j + n), K = k0+0..15 / +16..31
    v16h b0 = *(const v16h*)(brow + (size_t)0 * 16 * CC + k0 + hi * 16);
    v16h b1 = *(const v16h*)(brow + (size_t)1 * 16 * CC + k0 + hi * 16);
    v16h b2 = *(const v16h*)(brow + (size_t)2 * 16 * CC + k0 + hi * 16);
    v16h b3 = *(const v16h*)(brow + (size_t)3 * 16 * CC + k0 + hi * 16);
    acc0 = __builtin_amdgcn_wmma_f32_16x16x32_f16(false, a, false, b0,
                                                  (short)0, acc0, false, false);
    acc1 = __builtin_amdgcn_wmma_f32_16x16x32_f16(false, a, false, b1,
                                                  (short)0, acc1, false, false);
    acc2 = __builtin_amdgcn_wmma_f32_16x16x32_f16(false, a, false, b2,
                                                  (short)0, acc2, false, false);
    acc3 = __builtin_amdgcn_wmma_f32_16x16x32_f16(false, a, false, b3,
                                                  (short)0, acc3, false, false);
  }
  // D 16x16 f32: lane holds N=lm, VGPR r holds M = r + 8*hi
  float* out = sim + (size_t)b * NN * NN + (size_t)(row0 + hi * 8) * NN
             + col0 + lm;
#pragma unroll
  for (int r = 0; r < 8; ++r) {
    out[(size_t)r * NN + 0]  = acc0[r];
    out[(size_t)r * NN + 16] = acc1[r];
    out[(size_t)r * NN + 32] = acc2[r];
    out[(size_t)r * NN + 48] = acc3[r];
  }
}

// ---------------------------------------------------------------------------
// K3: per row of sim: top-4 (vals+idx), streaming-logsumexp entropy,
// gather pos_B, scatter disp into (b,k,c,H,W).
// ---------------------------------------------------------------------------
__global__ __launch_bounds__(256) void row_pass_kernel(
    const float* __restrict__ sim, const float* __restrict__ posA,
    const float* __restrict__ posB, float* __restrict__ ent_out,
    float* __restrict__ vals_ws, float* __restrict__ tpos_ws,
    float* __restrict__ disp_ws) {
  const int row = blockIdx.x;          // 0 .. B*N-1
  const int b   = row >> 12;
  const int n   = row & (NN - 1);
  const int t   = threadIdx.x;
  const float* srow = sim + (size_t)row * NN;

  float v[16];
#pragma unroll
  for (int i = 0; i < 16; ++i) v[i] = srow[t + i * 256];

  // per-thread sorted-descending top-4 + streaming softmax stats (t_i = 20*s)
  float tv[4] = {-1e30f, -1e30f, -1e30f, -1e30f};
  int   ti[4] = {0, 0, 0, 0};
  float m = -1e30f, l = 0.0f, w = 0.0f;
#pragma unroll
  for (int i = 0; i < 16; ++i) {
    float x = v[i];
    int ix = t + i * 256;
    if (x > tv[3]) {
      int p = 3;
      while (p > 0 && x > tv[p - 1]) { tv[p] = tv[p - 1]; ti[p] = ti[p - 1]; --p; }
      tv[p] = x; ti[p] = ix;
    }
    float s = x * INV_T;
    if (s > m) { float e = __expf(m - s); l *= e; w *= e; m = s; }
    float e = __expf(s - m);
    l += e; w += e * s;
  }

  __shared__ float sm[256], sl[256], sw[256];
  __shared__ float stv[256][4];
  __shared__ int   sti[256][4];
  sm[t] = m; sl[t] = l; sw[t] = w;
#pragma unroll
  for (int k = 0; k < 4; ++k) { stv[t][k] = tv[k]; sti[t][k] = ti[k]; }
  __syncthreads();

  for (int stride = 128; stride > 0; stride >>= 1) {
    if (t < stride) {
      // merge softmax stats
      float m2 = sm[t + stride], l2 = sl[t + stride], w2 = sw[t + stride];
      float M  = fmaxf(sm[t], m2);
      float e1 = __expf(sm[t] - M), e2 = __expf(m2 - M);
      sl[t] = sl[t] * e1 + l2 * e2;
      sw[t] = sw[t] * e1 + w2 * e2;
      sm[t] = M;
      // merge two sorted-descending 4-lists
      float av[4], bv[4]; int ai[4], bi[4];
#pragma unroll
      for (int k = 0; k < 4; ++k) {
        av[k] = stv[t][k];          ai[k] = sti[t][k];
        bv[k] = stv[t + stride][k]; bi[k] = sti[t + stride][k];
      }
      int p = 0, q = 0;
#pragma unroll
      for (int k = 0; k < 4; ++k) {
        if (av[p] >= bv[q]) { stv[t][k] = av[p]; sti[t][k] = ai[p]; ++p; }
        else               { stv[t][k] = bv[q]; sti[t][k] = bi[q]; ++q; }
      }
    }
    __syncthreads();
  }

  if (t == 0) {
    float M = sm[0], L = sl[0], Wv = sw[0];
    // entropy = lse - E[t] = (M + log L) - Wv/L
    ent_out[row] = M + __logf(L) - Wv / L;

    float pax = posA[(size_t)row * 2 + 0];
    float pay = posA[(size_t)row * 2 + 1];
    const int h = n >> 6, x = n & 63;
#pragma unroll
    for (int k = 0; k < 4; ++k) {
      vals_ws[row * 4 + k] = stv[0][k];
      int id = sti[0][k];
      float px = posB[((size_t)b * NN + id) * 2 + 0];
      float py = posB[((size_t)b * NN + id) * 2 + 1];
      tpos_ws[(row * 4 + k) * 2 + 0] = px;
      tpos_ws[(row * 4 + k) * 2 + 1] = py;
      disp_ws[((((b * 4 + k) * 2 + 0) * HH + h) * WW) + x] = px - pax;
      disp_ws[((((b * 4 + k) * 2 + 1) * HH + h) * WW) + x] = py - pay;
    }
  }
}

// ---------------------------------------------------------------------------
// K4: 7x7 window variance (both coords, n = 98), geo term, K=4 softmax,
// refined_warp.
// ---------------------------------------------------------------------------
__global__ __launch_bounds__(256) void finalize_kernel(
    const float* __restrict__ disp_ws, const float* __restrict__ vals_ws,
    const float* __restrict__ tpos_ws, float* __restrict__ warp_out) {
  int row = blockIdx.x * blockDim.x + threadIdx.x;
  if (row >= BB * NN) return;
  const int b = row >> 12, n = row & (NN - 1);
  const int h = n >> 6, x0 = n & 63;

  float cmb[4];
#pragma unroll
  for (int k = 0; k < 4; ++k) {
    float s1 = 0.0f, s2 = 0.0f;
#pragma unroll
    for (int c = 0; c < 2; ++c) {
      const float* base = disp_ws + (size_t)(((b * 4 + k) * 2 + c) * HH) * WW;
      for (int dy = -3; dy <= 3; ++dy) {
        int y = h + dy; if (y < 0 || y >= HH) continue;
        for (int dx = -3; dx <= 3; ++dx) {
          int x = x0 + dx; if (x < 0 || x >= WW) continue;
          float d = base[y * WW + x];
          s1 += d; s2 += d * d;
        }
      }
    }
    float var = (s2 - s1 * s1 * (1.0f / 98.0f)) * (1.0f / 97.0f);
    float geo = 1.0f / (1.0f + var * 100.0f);
    cmb[k] = vals_ws[row * 4 + k] + 0.5f * geo;
  }
  float m = fmaxf(fmaxf(cmb[0], cmb[1]), fmaxf(cmb[2], cmb[3]));
  float e[4], s = 0.0f;
#pragma unroll
  for (int k = 0; k < 4; ++k) { e[k] = __expf((cmb[k] - m) * INV_T); s += e[k]; }
  float inv = 1.0f / s, rx = 0.0f, ry = 0.0f;
#pragma unroll
  for (int k = 0; k < 4; ++k) {
    float wk = e[k] * inv;
    rx += tpos_ws[(row * 4 + k) * 2 + 0] * wk;
    ry += tpos_ws[(row * 4 + k) * 2 + 1] * wk;
  }
  warp_out[row * 2 + 0] = rx;
  warp_out[row * 2 + 1] = ry;
}

// ---------------------------------------------------------------------------
extern "C" void kernel_launch(void* const* d_in, const int* in_sizes, int n_in,
                              void* d_out, int out_size, void* d_ws,
                              size_t ws_size, hipStream_t stream) {
  const float* featA = (const float*)d_in[0];
  const float* featB = (const float*)d_in[1];
  const float* posA  = (const float*)d_in[2];
  const float* posB  = (const float*)d_in[3];
  // d_in[4], d_in[5] = H, W scalars (compile-time constants here)

  float* out  = (float*)d_out;
  float* warp = out;                 // (B,N,2)  = 32768 floats
  float* ent  = out + BB * NN * 2;   // (B,N)    = 16384 floats
  float* sim  = out + BB * NN * 3;   // (B,N,N)  = 67108864 floats

  char* ws = (char*)d_ws;
  _Float16* Ah  = (_Float16*)(ws);                                 // 4 MB
  _Float16* Bh  = (_Float16*)(ws + ((size_t)4 << 20));             // 4 MB
  float* vals   = (float*)(ws + ((size_t)8 << 20));                // 256 KB
  float* tpos   = (float*)(ws + ((size_t)8 << 20) + (256u << 10)); // 512 KB
  float* disp   = (float*)(ws + ((size_t)8 << 20) + (768u << 10)); // 512 KB

  const int rows = BB * NN;  // 16384

  norm_cvt_kernel<<<dim3(rows / 8), 256, 0, stream>>>(featA, Ah, rows);
  norm_cvt_kernel<<<dim3(rows / 8), 256, 0, stream>>>(featB, Bh, rows);

  // 64x64 output tile per block (4 waves x 16x64 strips)
  sim_gemm_kernel<<<dim3(NN / 64, NN / 64, BB), 128, 0, stream>>>(Ah, Bh, sim);

  row_pass_kernel<<<dim3(rows), 256, 0, stream>>>(sim, posA, posB, ent, vals,
                                                  tpos, disp);

  finalize_kernel<<<dim3((rows + 255) / 256), 256, 0, stream>>>(disp, vals,
                                                                tpos, warp);
}
